// Plane_refine_block_41927470743686
// MI455X (gfx1250) — compile-verified, matrix operand order checked by reading, not run on hardware
//
#include <hip/hip_runtime.h>
#include <hip/hip_bf16.h>

typedef __attribute__((ext_vector_type(16))) _Float16 v16h;
typedef __attribute__((ext_vector_type(2)))  _Float16 v2h;
typedef __attribute__((ext_vector_type(8)))  float    v8f;

constexpr int N = 262144;
constexpr int P = 128;
constexpr int D = 64;
constexpr int SENT = (int)0xFF800000u;   // -inf bits; ReLU outputs (bits >= 0) beat this as int
constexpr int CHUNKS = 32;               // mask/pool chunks along N

// Fragment-order helpers (16-bit A/B WMMA layout, wave32):
//   lane = hi*16 + (row|col within tile), half j of k-local kl:
//   hi = (kl>>3)&1 ; j = (kl&7) | ((kl>>4)<<3)
__device__ __forceinline__ void frag_pos(int kl, int& hi, int& j) {
    hi = (kl >> 3) & 1;
    j  = (kl & 7) | ((kl >> 4) << 3);
}

// ---------------------------------------------------------------------------
// Kernel 1: fused MLP  f=[feature|feature_geo] -> h1 -> h2 -> scores
// 256 threads = 8 waves; each wave computes a 16-point tile with
// v_wmma_f32_16x16x32_f16. Weights pre-swizzled to fragment order in LDS so
// every fragment is one contiguous 32B LDS read per lane.
// ---------------------------------------------------------------------------
__global__ __launch_bounds__(256) void mlp_wmma_kernel(
    const float* __restrict__ feature, const float* __restrict__ feature_geo,
    const float* __restrict__ w1, const float* __restrict__ b1,
    const float* __restrict__ g1, const float* __restrict__ be1,
    const float* __restrict__ m1, const float* __restrict__ v1,
    const float* __restrict__ w2, const float* __restrict__ b2,
    const float* __restrict__ g2, const float* __restrict__ be2,
    const float* __restrict__ m2, const float* __restrict__ v2,
    const float* __restrict__ w3, const float* __restrict__ b3,
    _Float16* __restrict__ h2_out, float* __restrict__ scores_out)
{
    __shared__ v16h w1f[16 * 32];          // 16 KB: 16 frags (kt*4+nt) x 32 lanes
    __shared__ v16h w2f[8 * 32];           //  8 KB:  8 frags (kt*4+nt) x 32 lanes
    __shared__ v16h h1f[8 * 2 * 32];       // 16 KB: per-wave A-frag staging / h2 bounce
    __shared__ float A1s[64], C1s[64], A2s[64], C2s[64], w3s[64];

    _Float16* w1fh = (_Float16*)w1f;
    _Float16* w2fh = (_Float16*)w2f;
    _Float16* h1fh = (_Float16*)h1f;

    const int tid = threadIdx.x;

    // ---- Stage weights into fragment order (one-time, scattered 2B LDS) ----
    for (int i = tid; i < 128 * 64; i += 256) {
        int k = i >> 6, n = i & 63;
        int kt = k >> 5, kl = k & 31, hi, j;
        frag_pos(kl, hi, j);
        int frag = kt * 4 + (n >> 4);
        w1fh[(frag * 32 + hi * 16 + (n & 15)) * 16 + j] = (_Float16)w1[i];
    }
    for (int i = tid; i < 64 * 64; i += 256) {
        int k = i >> 6, n = i & 63;
        int kt = k >> 5, kl = k & 31, hi, j;
        frag_pos(kl, hi, j);
        int frag = kt * 4 + (n >> 4);
        w2fh[(frag * 32 + hi * 16 + (n & 15)) * 16 + j] = (_Float16)w2[i];
    }
    if (tid < 64) {
        float s1 = g1[tid] * rsqrtf(v1[tid] + 1e-5f);
        A1s[tid] = s1;
        C1s[tid] = (b1[tid] - m1[tid]) * s1 + be1[tid];
        float s2 = g2[tid] * rsqrtf(v2[tid] + 1e-5f);
        A2s[tid] = s2;
        C2s[tid] = (b2[tid] - m2[tid]) * s2 + be2[tid];
        w3s[tid] = w3[tid];
    }
    __syncthreads();

    const int lane = tid & 31;
    const int wave = tid >> 5;
    const int mrow = lane & 15;            // M row this lane owns (A layout)
    const int kg   = (lane >> 4) * 8;      // K group base per half-wave
    const int base = blockIdx.x * 128 + wave * 16;
    const int m    = base + mrow;

    // ---- A fragments for GEMM1 from global (float4 runs -> f16) ----
    const float* frow = feature     + (size_t)m * 64;
    const float* grow = feature_geo + (size_t)m * 64;
    v16h a1[4];
#pragma unroll
    for (int kt = 0; kt < 4; ++kt) {
        const float* src = (kt < 2) ? (frow + kt * 32) : (grow + (kt - 2) * 32);
        float4 p0 = *(const float4*)(src + kg);
        float4 p1 = *(const float4*)(src + kg + 4);
        float4 p2 = *(const float4*)(src + kg + 16);
        float4 p3 = *(const float4*)(src + kg + 20);
        float t[16] = {p0.x, p0.y, p0.z, p0.w, p1.x, p1.y, p1.z, p1.w,
                       p2.x, p2.y, p2.z, p2.w, p3.x, p3.y, p3.z, p3.w};
#pragma unroll
        for (int j = 0; j < 16; ++j) a1[kt][j] = (_Float16)t[j];
    }

    // ---- GEMM1: 16x128 * 128x64 -> 16x64 ----
    v8f acc[4];
#pragma unroll
    for (int nt = 0; nt < 4; ++nt) {
        v8f c = {};
#pragma unroll
        for (int kt = 0; kt < 4; ++kt) {
            v16h b = w1f[(kt * 4 + nt) * 32 + lane];
            c = __builtin_amdgcn_wmma_f32_16x16x32_f16(
                    false, a1[kt], false, b, (short)0, c, false, false);
        }
        acc[nt] = c;
    }

    // ---- BN + ReLU -> h1 directly in A-fragment order (per-wave region) ----
#pragma unroll
    for (int nt = 0; nt < 4; ++nt) {
        int col = nt * 16 + mrow;
        int kt2 = col >> 5, kl = col & 31, hi, j;
        frag_pos(kl, hi, j);
        float sA = A1s[col], sC = C1s[col];
#pragma unroll
        for (int r = 0; r < 8; ++r) {
            int row = (lane >> 4) * 8 + r;
            float h = fmaxf(acc[nt][r] * sA + sC, 0.0f);
            h1fh[((wave * 2 + kt2) * 32 + hi * 16 + row) * 16 + j] = (_Float16)h;
        }
    }
    __syncthreads();

    // ---- A fragments for GEMM2: contiguous 32B LDS reads ----
    v16h a2[2];
#pragma unroll
    for (int kt = 0; kt < 2; ++kt) a2[kt] = h1f[(wave * 2 + kt) * 32 + lane];

    // ---- GEMM2: 16x64 * 64x64 -> 16x64 ----
#pragma unroll
    for (int nt = 0; nt < 4; ++nt) {
        v8f c = {};
#pragma unroll
        for (int kt = 0; kt < 2; ++kt) {
            v16h b = w2f[(kt * 4 + nt) * 32 + lane];
            c = __builtin_amdgcn_wmma_f32_16x16x32_f16(
                    false, a2[kt], false, b, (short)0, c, false, false);
        }
        acc[nt] = c;
    }
    __syncthreads();   // h1f region about to be reused as row-major h2 bounce

    // ---- BN + ReLU -> h2 (row-major LDS bounce) + fc3 partial dot ----
    _Float16* h2s = h1fh + wave * 1024;    // 16 rows x 64 halves = 2KB per wave
    float spart[8] = {0, 0, 0, 0, 0, 0, 0, 0};
#pragma unroll
    for (int nt = 0; nt < 4; ++nt) {
        int col = nt * 16 + mrow;
        float sA = A2s[col], sC = C2s[col], wv = w3s[col];
#pragma unroll
        for (int r = 0; r < 8; ++r) {
            int row = (lane >> 4) * 8 + r;
            float h = fmaxf(acc[nt][r] * sA + sC, 0.0f);
            h2s[row * 64 + col] = (_Float16)h;
            spart[r] += h * wv;
        }
    }
    float b3v = b3[0];
#pragma unroll
    for (int r = 0; r < 8; ++r) {
        float s = spart[r];
        s += __shfl_xor(s, 1);
        s += __shfl_xor(s, 2);
        s += __shfl_xor(s, 4);
        s += __shfl_xor(s, 8);
        if (mrow == 0) scores_out[base + (lane >> 4) * 8 + r] = fmaxf(s + b3v, 0.0f);
    }

    // ---- flush h2 tile: 64B per lane, b128 stores (full cache lines) ----
    const uint4* sv = (const uint4*)h2s;
    uint4* dv = (uint4*)(h2_out + (size_t)base * 64);
#pragma unroll
    for (int q = 0; q < 4; ++q) dv[lane * 4 + q] = sv[lane * 4 + q];
}

// ---------------------------------------------------------------------------
// Kernel 0: re-seed global pooling accumulators to -inf sentinel each launch.
// ---------------------------------------------------------------------------
__global__ void init_acc_kernel(int* __restrict__ g_acc)
{
    int i = blockIdx.x * blockDim.x + threadIdx.x;
    if (i < 2 * P * D) g_acc[i] = SENT;
}

// ---------------------------------------------------------------------------
// Kernel 2: per (plane, chunk): 4 points/thread mask math with float4 streams;
// masked max-pool via wave32 ballot + cooperative 128B h2-row reads, running
// maxima in VGPRs (lane owns dims 2*lane, 2*lane+1); LDS/global atomics only
// at block exit.
// ---------------------------------------------------------------------------
__global__ __launch_bounds__(256) void mask_pool_kernel(
    const float* __restrict__ xyz, const float* __restrict__ centers,
    const float* __restrict__ plane_center, const float* __restrict__ plane_normal,
    const float* __restrict__ pmin, const float* __restrict__ pmax,
    const float* __restrict__ scores, const _Float16* __restrict__ h2,
    float* __restrict__ out, int* __restrict__ g_acc)
{
    const int p     = blockIdx.y;
    const int chunk = blockIdx.x;
    const int tid   = threadIdx.x;
    const int lane  = tid & 31;

    __shared__ int lacc[2 * D];
    if (tid < 2 * D) lacc[tid] = SENT;
    __syncthreads();

    const float cx = centers[0], cy = centers[1], cz = centers[2];
    const float nx = plane_normal[p * 3 + 0];
    const float ny = plane_normal[p * 3 + 1];
    const float nz = plane_normal[p * 3 + 2];
    const float off = plane_center[p * 3 + 0] * nx +
                      plane_center[p * 3 + 1] * ny +
                      plane_center[p * 3 + 2] * nz;
    const float mn0 = pmin[p * 3 + 0], mx0 = pmax[p * 3 + 0];
    const float mn1 = pmin[p * 3 + 1], mx1 = pmax[p * 3 + 1];
    const float mn2 = pmin[p * 3 + 2], mx2 = pmax[p * 3 + 2];

    float* out_sm   = out;
    float* out_mask = out + (size_t)1 * P * N;
    float* out_on   = out + (size_t)2 * P * N;
    float* out_off  = out + (size_t)3 * P * N;

    float von0 = __int_as_float(SENT), von1 = __int_as_float(SENT);
    float vof0 = __int_as_float(SENT), vof1 = __int_as_float(SENT);

    const int span = N / CHUNKS;                 // 8192 (exact multiple of 1024)
    const int n0   = chunk * span;

    for (int i0 = tid * 4; i0 < span; i0 += 1024) {
        int n = n0 + i0;
        const float4* xv = (const float4*)(xyz + (size_t)n * 3);
        float4 q0 = xv[0], q1 = xv[1], q2 = xv[2];
        float fx[12] = {q0.x, q0.y, q0.z, q0.w, q1.x, q1.y,
                        q1.z, q1.w, q2.x, q2.y, q2.z, q2.w};
        float4 scv = *(const float4*)(scores + n);
        float sc[4] = {scv.x, scv.y, scv.z, scv.w};

        float r_sm[4], r_mk[4], r_on[4], r_of[4];
        bool onb[4], ofb[4];
#pragma unroll
        for (int s = 0; s < 4; ++s) {
            float x = fx[3 * s + 0] + cx;
            float y = fx[3 * s + 1] + cy;
            float z = fx[3 * s + 2] + cz;
            bool ok = ((mx0 == 0.0f) || (x >= mn0 && x < mx0)) &&
                      ((mx1 == 0.0f) || (y >= mn1 && y < mx1)) &&
                      ((mx2 == 0.0f) || (z >= mn2 && z < mx2));
            float dist = fabsf(x * nx + y * ny + z * nz - off);
            bool msk = ok && (dist < 0.1f);
            bool on  = msk && (sc[s] > 0.0f);    // sigmoid(s) > 0.5 <=> s > 0
            onb[s] = on;
            ofb[s] = msk && !on;
            r_sm[s] = msk ? sc[s] : 0.0f;
            r_mk[s] = msk ? 1.0f : 0.0f;
            r_on[s] = on ? 1.0f : 0.0f;
            r_of[s] = ofb[s] ? 1.0f : 0.0f;
        }
        size_t idx = (size_t)p * N + n;
        *(float4*)(out_sm   + idx) = make_float4(r_sm[0], r_sm[1], r_sm[2], r_sm[3]);
        *(float4*)(out_mask + idx) = make_float4(r_mk[0], r_mk[1], r_mk[2], r_mk[3]);
        *(float4*)(out_on   + idx) = make_float4(r_on[0], r_on[1], r_on[2], r_on[3]);
        *(float4*)(out_off  + idx) = make_float4(r_of[0], r_of[1], r_of[2], r_of[3]);

        // Cooperative pooling: lane b's sub-point s lives at itbase + 4*b + s.
        int itbase = n - lane * 4;               // uniform across the wave
#pragma unroll
        for (int s = 0; s < 4; ++s) {
            unsigned int bo = (unsigned int)__ballot(onb[s]);
            while (bo) {
                int b = __ffs(bo) - 1; bo &= bo - 1;
                v2h v = *(const v2h*)(h2 + (size_t)(itbase + 4 * b + s) * 64 + 2 * lane);
                von0 = fmaxf(von0, (float)v[0]);
                von1 = fmaxf(von1, (float)v[1]);
            }
            unsigned int bf = (unsigned int)__ballot(ofb[s]);
            while (bf) {
                int b = __ffs(bf) - 1; bf &= bf - 1;
                v2h v = *(const v2h*)(h2 + (size_t)(itbase + 4 * b + s) * 64 + 2 * lane);
                vof0 = fmaxf(vof0, (float)v[0]);
                vof1 = fmaxf(vof1, (float)v[1]);
            }
        }
    }

    // Merge per-lane maxima (dims 2*lane, 2*lane+1) across the block's waves.
    atomicMax(&lacc[2 * lane + 0],     __float_as_int(von0));
    atomicMax(&lacc[2 * lane + 1],     __float_as_int(von1));
    atomicMax(&lacc[D + 2 * lane + 0], __float_as_int(vof0));
    atomicMax(&lacc[D + 2 * lane + 1], __float_as_int(vof1));
    __syncthreads();

    if (tid < 2 * D) {
        int v = lacc[tid];
        if (v != SENT) {
            int which = tid >> 6, d = tid & 63;
            atomicMax(&g_acc[(which * P + p) * D + d], v);
        }
    }
}

// ---------------------------------------------------------------------------
// Kernel 3: accumulators -> on_feat/off_feat (sentinel => 0.0).
// ---------------------------------------------------------------------------
__global__ void pool_fixup_kernel(const int* __restrict__ g_acc, float* __restrict__ out)
{
    int i = blockIdx.x * blockDim.x + threadIdx.x;
    if (i < 2 * P * D) {
        int v = g_acc[i];
        out[(size_t)4 * P * N + i] = (v == SENT) ? 0.0f : __int_as_float(v);
    }
}

extern "C" void kernel_launch(void* const* d_in, const int* in_sizes, int n_in,
                              void* d_out, int out_size, void* d_ws, size_t ws_size,
                              hipStream_t stream)
{
    const float* feature      = (const float*)d_in[0];
    const float* feature_geo  = (const float*)d_in[1];
    const float* xyz          = (const float*)d_in[2];
    const float* centers      = (const float*)d_in[3];
    const float* plane_center = (const float*)d_in[4];
    const float* plane_normal = (const float*)d_in[5];
    const float* pmin         = (const float*)d_in[6];
    const float* pmax         = (const float*)d_in[7];
    const float* w1 = (const float*)d_in[8];
    const float* b1 = (const float*)d_in[9];
    const float* g1 = (const float*)d_in[10];
    const float* be1 = (const float*)d_in[11];
    const float* m1 = (const float*)d_in[12];
    const float* v1 = (const float*)d_in[13];
    const float* w2 = (const float*)d_in[14];
    const float* b2 = (const float*)d_in[15];
    const float* g2 = (const float*)d_in[16];
    const float* be2 = (const float*)d_in[17];
    const float* m2 = (const float*)d_in[18];
    const float* v2 = (const float*)d_in[19];
    const float* w3 = (const float*)d_in[20];
    const float* b3 = (const float*)d_in[21];

    // Workspace layout: h2 f16 [N*64] | scores f32 [N] | g_acc int [2*P*D]
    char* ws = (char*)d_ws;
    _Float16* h2     = (_Float16*)ws;
    float*    scores = (float*)(ws + (size_t)N * D * sizeof(_Float16));
    int*      g_acc  = (int*)(ws + (size_t)N * D * sizeof(_Float16) + (size_t)N * sizeof(float));
    float*    out    = (float*)d_out;

    init_acc_kernel<<<(2 * P * D + 255) / 256, 256, 0, stream>>>(g_acc);

    mlp_wmma_kernel<<<N / 128, 256, 0, stream>>>(
        feature, feature_geo, w1, b1, g1, be1, m1, v1,
        w2, b2, g2, be2, m2, v2, w3, b3, h2, scores);

    dim3 grid(CHUNKS, P);
    mask_pool_kernel<<<grid, 256, 0, stream>>>(
        xyz, centers, plane_center, plane_normal, pmin, pmax,
        scores, h2, out, g_acc);

    pool_fixup_kernel<<<(2 * P * D + 255) / 256, 256, 0, stream>>>(g_acc, out);
}
